// Attention_28235115003955
// MI455X (gfx1250) — compile-verified, matrix operand order checked by reading, not run on hardware
//
#include <hip/hip_runtime.h>

// ---------------------------------------------------------------------------
// Fused multi-head attention for MI455X (gfx1250, wave32, WMMA bf16 path)
//   B=8, S=2048, D=512, H=8, Dh=64
// Phase 0: Wq/Wk/Wv fp32 [K,N] -> bf16 N-major WT[3][N][K] (one-time, 3MB)
// Phase 1: projections with LDS-staged bf16 X tiles; 8 WMMAs per 32-k chunk
//          per wave (2 row-groups x 4 col-tiles). qw,kw -> [B,H,S,64] bf16;
//          vw -> transposed [B,H,64,S] bf16.
// Phase 2: flash attention, 32 query rows per wave, online softmax,
//          K/V fragments reused across two query row-groups.
// ---------------------------------------------------------------------------

typedef __attribute__((ext_vector_type(16))) __bf16 v16bf;
typedef __attribute__((ext_vector_type(8)))  __bf16 v8bf;
typedef __attribute__((ext_vector_type(8)))  float  v8f;
typedef __attribute__((ext_vector_type(4)))  float  v4f;

#define NB       8
#define NS       2048
#define ND       512
#define NH       8
#define NDH      64
#define MASK_NEG 1e10f

__device__ __forceinline__ v8f wmma_bf16(v16bf a, v16bf b, v8f c) {
  return __builtin_amdgcn_wmma_f32_16x16x32_bf16(
      /*neg_a=*/false, a, /*neg_b=*/false, b,
      /*c_mod=*/(short)0, c, /*reuse_a=*/false, /*reuse_b=*/false);
}

__device__ __forceinline__ v16bf frag_from(const __bf16* p, int kbase) {
  v8bf lo = *(const v8bf*)(p + kbase);
  v8bf hi = *(const v8bf*)(p + kbase + 16);
  v16bf f;
#pragma unroll
  for (int i = 0; i < 8; ++i) { f[i] = lo[i]; f[8 + i] = hi[i]; }
  return f;
}

// ---------------------------------------------------------------------------
// Phase 0: weight convert + transpose  (fp32 [K,N] -> bf16 [which][N][K])
// grid = (512*512/256, 3), block = 256
// ---------------------------------------------------------------------------
__global__ __launch_bounds__(256) void wconv_kernel(
    const float* __restrict__ Wq, const float* __restrict__ Wk,
    const float* __restrict__ Wv, __bf16* __restrict__ WT)
{
  const int which = blockIdx.y;
  const float* W = (which == 0) ? Wq : ((which == 1) ? Wk : Wv);
  const int idx = blockIdx.x * 256 + threadIdx.x;   // linear over [K,N], N fastest
  const int k = idx >> 9;
  const int n = idx & 511;
  WT[((size_t)which * ND + n) * ND + k] = (__bf16)W[idx];
}

// ---------------------------------------------------------------------------
// Phase 1: Y = X @ W.  Block = 8 waves; stages 32x512 X tile as bf16 in LDS.
// Wave w owns output cols [w*64, w*64+64) x 32 rows -> 8 accum tiles.
// grid = (M/32, 3), block = 256
// ---------------------------------------------------------------------------
__global__ __launch_bounds__(256) void proj_kernel(
    const float* __restrict__ Xq, const float* __restrict__ Xk,
    const float* __restrict__ Xv, const __bf16* __restrict__ WT,
    __bf16* __restrict__ qw, __bf16* __restrict__ kw,
    __bf16* __restrict__ vwT)
{
  __shared__ __attribute__((aligned(16))) __bf16 xs[32 * ND];   // 32 KB

  const int tid   = threadIdx.x;
  const int wave  = tid >> 5;
  const int lane  = tid & 31;
  const int mt    = blockIdx.x;     // 0..511  (M = 16384, 32 rows/block)
  const int which = blockIdx.y;     // 0:q 1:k 2:v

  const float* X = (which == 0) ? Xq : ((which == 1) ? Xk : Xv);
  const __bf16* W = WT + (size_t)which * ND * ND;  // [N][K] bf16

  const int m0 = mt * 32;

  // ---- cooperative X stage: 32x512 fp32 -> bf16 LDS, coalesced b128 loads ----
#pragma unroll
  for (int i = 0; i < 8; ++i) {
    const int c   = tid + i * 256;          // chunk of 8 floats, 2048 chunks
    const int row = c >> 6;                 // 64 chunks per row
    const int col = (c & 63) * 8;
    const float* xp = X + (size_t)(m0 + row) * ND + col;
    v4f f0 = *(const v4f*)(xp);
    v4f f1 = *(const v4f*)(xp + 4);
    v8bf hchunk;
#pragma unroll
    for (int j = 0; j < 4; ++j) { hchunk[j] = (__bf16)f0[j]; hchunk[4 + j] = (__bf16)f1[j]; }
    *(v8bf*)(&xs[row * ND + col]) = hchunk;
  }
  __syncthreads();

  const int l15   = lane & 15;
  const int half  = lane >> 4;
  const int kbase = half * 8;
  const int n0    = wave * 64;              // 4 col-tiles per wave

  v8f acc[2][4] = {};
  for (int kk = 0; kk < ND; kk += 32) {
    v16bf a[2];
#pragma unroll
    for (int g = 0; g < 2; ++g)
      a[g] = frag_from(&xs[(g * 16 + l15) * ND + kk], kbase);
#pragma unroll
    for (int t = 0; t < 4; ++t) {
      v16bf b = frag_from(W + (size_t)(n0 + t * 16 + l15) * ND + kk, kbase);
      acc[0][t] = wmma_bf16(a[0], b, acc[0][t]);
      acc[1][t] = wmma_bf16(a[1], b, acc[1][t]);
    }
  }

  // ---- epilogue ----
  if (which == 2) {
    // vwT[bh][dh][s]: per lane r=0..7 are consecutive s -> packed b128 store
#pragma unroll
    for (int g = 0; g < 2; ++g) {
#pragma unroll
      for (int t = 0; t < 4; ++t) {
        const int n  = n0 + t * 16 + l15;
        const int h  = n >> 6, dh = n & 63;
        const int m  = m0 + g * 16 + 8 * half;   // first of 8 consecutive rows
        const int bb = m >> 11, s = m & 2047;
        const size_t bh = (size_t)bb * NH + h;
        v8bf pk;
#pragma unroll
        for (int r = 0; r < 8; ++r) pk[r] = (__bf16)acc[g][t][r];
        *(v8bf*)(&vwT[(bh * NDH + dh) * NS + s]) = pk;
      }
    }
  } else {
    __bf16* dst = (which == 0) ? qw : kw;
#pragma unroll
    for (int g = 0; g < 2; ++g) {
#pragma unroll
      for (int t = 0; t < 4; ++t) {
        const int n  = n0 + t * 16 + l15;
        const int h  = n >> 6, dh = n & 63;
#pragma unroll
        for (int r = 0; r < 8; ++r) {
          const int m  = m0 + g * 16 + r + 8 * half;
          const int bb = m >> 11, s = m & 2047;
          const size_t bh = (size_t)bb * NH + h;
          dst[(bh * NS + s) * NDH + dh] = (__bf16)acc[g][t][r];
        }
      }
    }
  }
}

// ---------------------------------------------------------------------------
// Phase 2: flash attention. One wave owns 32 query rows (2 groups of 16),
// streams keys in steps of 32; K/V fragments reused across both groups.
// grid = (S/32, H, B), block = 32
// ---------------------------------------------------------------------------
__global__ __launch_bounds__(32) void attn_kernel(
    const __bf16* __restrict__ qw, const __bf16* __restrict__ kw,
    const __bf16* __restrict__ vwT,
    const float* __restrict__ v_mask, const float* __restrict__ q_mask,
    float* __restrict__ out)
{
  __shared__ __attribute__((aligned(16))) __bf16 pbuf[2][16 * 32]; // P staging

  const int lane  = threadIdx.x;
  const int qt    = blockIdx.x;       // 0..63
  const int h     = blockIdx.y;       // 0..7
  const int bb    = blockIdx.z;       // 0..7
  const int l15   = lane & 15;
  const int half  = lane >> 4;
  const int kbase = half * 8;
  const size_t bh = (size_t)bb * NH + h;
  const int q0    = qt * 32;

  // Q A-fragments: [row group][Dh chunk]
  v16bf qa[2][2];
#pragma unroll
  for (int g = 0; g < 2; ++g) {
    const __bf16* qrow = qw + (bh * NS + q0 + g * 16 + l15) * NDH;
#pragma unroll
    for (int t = 0; t < 2; ++t)
      qa[g][t] = frag_from(qrow + t * 32, kbase);
  }

  v8f o[2][4] = {};
  float mrun[2][8], lrun[2][8];
#pragma unroll
  for (int g = 0; g < 2; ++g)
#pragma unroll
    for (int r = 0; r < 8; ++r) { mrun[g][r] = -1e30f; lrun[g][r] = 0.0f; }

  const float* vm = v_mask + (size_t)bb * NS;

  for (int k0 = 0; k0 < NS; k0 += 32) {
    // ---- scores: s[g][kt] over keys k0+kt*16, reuse kb across groups ----
    v8f s[2][2] = {};
#pragma unroll
    for (int kt = 0; kt < 2; ++kt) {
      const __bf16* krow = kw + (bh * NS + k0 + kt * 16 + l15) * NDH;
#pragma unroll
      for (int t = 0; t < 2; ++t) {
        v16bf kb = frag_from(krow + t * 32, kbase);
        s[0][kt] = wmma_bf16(qa[0][t], kb, s[0][kt]);
        s[1][kt] = wmma_bf16(qa[1][t], kb, s[1][kt]);
      }
    }

    const float pen0 = (1.0f - vm[k0 + l15])      * MASK_NEG;
    const float pen1 = (1.0f - vm[k0 + 16 + l15]) * MASK_NEG;

    // ---- online softmax per row group ----
#pragma unroll
    for (int g = 0; g < 2; ++g) {
      float p0[8], p1[8];
#pragma unroll
      for (int r = 0; r < 8; ++r) {
        const float x0 = s[g][0][r] * 0.125f - pen0;   // 1/sqrt(64)
        const float x1 = s[g][1][r] * 0.125f - pen1;
        float mx = fmaxf(x0, x1);
        mx = fmaxf(mx, __shfl_xor(mx, 1));
        mx = fmaxf(mx, __shfl_xor(mx, 2));
        mx = fmaxf(mx, __shfl_xor(mx, 4));
        mx = fmaxf(mx, __shfl_xor(mx, 8));
        const float mnew  = fmaxf(mrun[g][r], mx);
        const float alpha = __expf(mrun[g][r] - mnew);
        p0[r] = __expf(x0 - mnew);
        p1[r] = __expf(x1 - mnew);
        float ps = p0[r] + p1[r];
        ps += __shfl_xor(ps, 1);
        ps += __shfl_xor(ps, 2);
        ps += __shfl_xor(ps, 4);
        ps += __shfl_xor(ps, 8);
        lrun[g][r] = lrun[g][r] * alpha + ps;
        mrun[g][r] = mnew;
        o[g][0][r] *= alpha; o[g][1][r] *= alpha;
        o[g][2][r] *= alpha; o[g][3][r] *= alpha;
      }
#pragma unroll
      for (int r = 0; r < 8; ++r) {
        const int row = r + 8 * half;
        pbuf[g][row * 32 + l15]      = (__bf16)p0[r];
        pbuf[g][row * 32 + 16 + l15] = (__bf16)p1[r];
      }
    }
    __builtin_amdgcn_wave_barrier();   // DS in-order per wave; block reordering
    v16bf pa[2];
#pragma unroll
    for (int g = 0; g < 2; ++g)
      pa[g] = frag_from(&pbuf[g][l15 * 32], kbase);
    __builtin_amdgcn_wave_barrier();

    // ---- O += P * V, reuse vb across groups ----
#pragma unroll
    for (int t = 0; t < 4; ++t) {
      v16bf vb = frag_from(vwT + (bh * NDH + t * 16 + l15) * NS + k0, kbase);
      o[0][t] = wmma_bf16(pa[0], vb, o[0][t]);
      o[1][t] = wmma_bf16(pa[1], vb, o[1][t]);
    }
  }

  // ---- epilogue: 1/l, query mask, fp32 store ----
#pragma unroll
  for (int g = 0; g < 2; ++g) {
#pragma unroll
    for (int r = 0; r < 8; ++r) {
      const int   m  = q0 + g * 16 + r + 8 * half;
      const float sc = (1.0f / lrun[g][r]) * q_mask[(size_t)bb * NS + m];
      const size_t base = ((size_t)bb * NS + m) * (NH * NDH) + h * NDH;
      out[base + 0  + l15] = o[g][0][r] * sc;
      out[base + 16 + l15] = o[g][1][r] * sc;
      out[base + 32 + l15] = o[g][2][r] * sc;
      out[base + 48 + l15] = o[g][3][r] * sc;
    }
  }
}

// ---------------------------------------------------------------------------
extern "C" void kernel_launch(void* const* d_in, const int* in_sizes, int n_in,
                              void* d_out, int out_size, void* d_ws, size_t ws_size,
                              hipStream_t stream) {
  (void)in_sizes; (void)n_in; (void)out_size; (void)ws_size;
  const float* q      = (const float*)d_in[0];
  const float* k      = (const float*)d_in[1];
  const float* v      = (const float*)d_in[2];
  const float* Wq     = (const float*)d_in[3];
  const float* Wk     = (const float*)d_in[4];
  const float* Wv     = (const float*)d_in[5];
  const float* v_mask = (const float*)d_in[6];
  const float* q_mask = (const float*)d_in[7];
  float* out = (float*)d_out;

  const size_t elems = (size_t)NB * NH * NS * NDH;   // 8.39M bf16 each
  __bf16* qw  = (__bf16*)d_ws;
  __bf16* kw  = qw + elems;
  __bf16* vwT = qw + 2 * elems;
  __bf16* WT  = qw + 3 * elems;                      // [3][512][512] bf16

  wconv_kernel<<<dim3((ND * ND) / 256, 3), dim3(256), 0, stream>>>(Wq, Wk, Wv, WT);

  proj_kernel<<<dim3((NB * NS) / 32, 3), dim3(256), 0, stream>>>(
      q, k, v, WT, qw, kw, vwT);

  attn_kernel<<<dim3(NS / 32, NH, NB), dim3(32), 0, stream>>>(
      qw, kw, vwT, v_mask, q_mask, out);
}
